// ConcatLoRALinear_80977313399763
// MI455X (gfx1250) — matrix operand order repaired and checked by reading, not compiled, not run on hardware
//
#include <hip/hip_runtime.h>

typedef __bf16 bf16_t;

#define MDIM 8192
#define NDIM 4096
#define KDIM 4096
#define NKB  (KDIM / 32)
#define SCALE_F 2.0f   /* alpha/r = 16/8 */

/* LDS layout (units: halves).  Row = 32 bf16 (64B) padded to 112B for bank spread
   while staying 16B-aligned for b128 access. */
#define LDSTR  56
#define XH_OFF 0
#define XL_OFF (128 * LDSTR)               /* 7168  */
#define WH_OFF (2 * 128 * LDSTR)           /* 14336 */
#define WL_OFF (WH_OFF + 256 * LDSTR)      /* 28672 */
#define STAGE  (WL_OFF + 256 * LDSTR)      /* 43008 halves = 86016 B */
#define LDS_BYTES (2 * STAGE * 2)          /* 172032 B (< 320 KB WGP LDS) */

#if defined(__gfx1250__) && __has_builtin(__builtin_amdgcn_global_load_async_to_lds_b128)
#define USE_ASYNC_LDS 1
#else
#define USE_ASYNC_LDS 0
#endif

#if defined(__HIP_DEVICE_COMPILE__)
typedef __attribute__((ext_vector_type(16))) __bf16 v16bf;
typedef __attribute__((ext_vector_type(8)))  __bf16 v8bf;
typedef __attribute__((ext_vector_type(4)))  __bf16 v4bf;
typedef __attribute__((ext_vector_type(8)))  float  v8f;
typedef __attribute__((ext_vector_type(4)))  float  v4f;
typedef __attribute__((ext_vector_type(4)))  int    v4i;

__device__ __forceinline__ void wait_async0() {
#if USE_ASYNC_LDS
#if __has_builtin(__builtin_amdgcn_s_wait_asynccnt)
    __builtin_amdgcn_s_wait_asynccnt(0);
#else
    asm volatile("s_wait_asynccnt 0x0" ::: "memory");
#endif
#endif
}
#endif /* device */

/* ------------------------------------------------------------------ */
/* Kernel 1: split x (fp32) into bf16 hi/lo pair (3xBF16 fp32 emulation) */
__global__ void splitx_kernel(const float* __restrict__ x,
                              bf16_t* __restrict__ xh, bf16_t* __restrict__ xl,
                              int n4) {
#if defined(__HIP_DEVICE_COMPILE__)
    int i = blockIdx.x * blockDim.x + threadIdx.x;
    if (i >= n4) return;
    v4f v = ((const v4f*)x)[i];
    v4bf h, l;
#pragma unroll
    for (int k = 0; k < 4; ++k) {
        float e = v[k];
        __bf16 hh = (__bf16)e;
        h[k] = hh;
        l[k] = (__bf16)(e - (float)hh);
    }
    ((v4bf*)xh)[i] = h;
    ((v4bf*)xl)[i] = l;
#endif
}

/* ------------------------------------------------------------------ */
/* Kernel 2: fold LoRA into the weight: W_eff[o,i] = W[o,i] + s*sum_{e,r} B[e,o,r]*A[e,r,i],
   then split to bf16 hi/lo.  64x64 output tile per block, rank-64 inner product from LDS. */
__global__ void weff_kernel(const float* __restrict__ W, const float* __restrict__ A,
                            const float* __restrict__ Bm,
                            bf16_t* __restrict__ wh, bf16_t* __restrict__ wl) {
#if defined(__HIP_DEVICE_COMPILE__)
    __shared__ float sA[64][65];   /* sA[er][ii] : A[e,r, i0+ii]  */
    __shared__ float sB[64][65];   /* sB[oo][er] : B[e, o0+oo, r] */
    const int t  = threadIdx.x;
    const int i0 = blockIdx.x * 64;
    const int o0 = blockIdx.y * 64;
    for (int idx = t; idx < 64 * 64; idx += 256) {
        int r0 = idx >> 6, c0 = idx & 63;
        sA[r0][c0] = A[(size_t)r0 * KDIM + i0 + c0];                 /* r0 = er */
        int e = c0 >> 3, rr = c0 & 7;                                 /* c0 = er */
        sB[r0][c0] = Bm[((size_t)e * NDIM + (o0 + r0)) * 8 + rr];     /* r0 = oo */
    }
    __syncthreads();
    const int oo = (t >> 4) << 2;
    const int ii = (t & 15) << 2;
    float acc[4][4] = {};
#pragma unroll 4
    for (int er = 0; er < 64; ++er) {
        float a0 = sA[er][ii + 0], a1 = sA[er][ii + 1];
        float a2 = sA[er][ii + 2], a3 = sA[er][ii + 3];
#pragma unroll
        for (int j = 0; j < 4; ++j) {
            float bv = sB[oo + j][er];
            acc[j][0] += bv * a0; acc[j][1] += bv * a1;
            acc[j][2] += bv * a2; acc[j][3] += bv * a3;
        }
    }
#pragma unroll
    for (int j = 0; j < 4; ++j)
#pragma unroll
        for (int k = 0; k < 4; ++k) {
            size_t o = (size_t)(o0 + oo + j), i = (size_t)(i0 + ii + k);
            float w = W[o * KDIM + i] + SCALE_F * acc[j][k];
            __bf16 hh = (__bf16)w;
            wh[o * KDIM + i] = hh;
            wl[o * KDIM + i] = (__bf16)(w - (float)hh);
        }
#endif
}

/* ------------------------------------------------------------------ */
/* Kernel 3: out = xsplit @ Weff_split^T + bias.  BM=128, BN=256, BK=32,
   8 waves (2x4), wave tile 64x64 = 4x4 WMMA frags, 3 bf16 passes (hi*hi+hi*lo+lo*hi). */
#if defined(__HIP_DEVICE_COMPILE__)
__device__ __forceinline__ void stage_addrs(int c, int kb, size_t m0, size_t n0,
                                            const bf16_t* xh, const bf16_t* xl,
                                            const bf16_t* wh, const bf16_t* wl,
                                            const bf16_t** g, int* l) {
    if (c < 1024) {               /* X: 128 rows * 4 chunks, hi then lo */
        const bf16_t* base = (c < 512) ? xh : xl;
        int lbase = (c < 512) ? XH_OFF : XL_OFF;
        int cc = c & 511, row = cc >> 2, p = cc & 3;
        *g = base + (m0 + row) * (size_t)KDIM + (size_t)kb * 32 + p * 8;
        *l = lbase + row * LDSTR + p * 8;
    } else {                      /* W: 256 rows * 4 chunks, hi then lo */
        const bf16_t* base = (c < 2048) ? wh : wl;
        int lbase = (c < 2048) ? WH_OFF : WL_OFF;
        int cc = (c - 1024) & 1023, row = cc >> 2, p = cc & 3;
        *g = base + (n0 + row) * (size_t)KDIM + (size_t)kb * 32 + p * 8;
        *l = lbase + row * LDSTR + p * 8;
    }
}

__device__ __forceinline__ v16bf mkfrag(const bf16_t* p0, const bf16_t* p1) {
    union { v16bf v; v8bf h[2]; } u;
    u.h[0] = *(const v8bf*)p0;    /* ds_load_b128 */
    u.h[1] = *(const v8bf*)p1;    /* ds_load_b128 */
    return u.v;
}

__device__ __forceinline__ void compute_stage(const bf16_t* S, v8f c[4][4],
                                              int waveM, int waveN, int lhalf, int lsel) {
    v16bf ah[4], al[4];
#pragma unroll
    for (int mi = 0; mi < 4; ++mi) {
        int row = waveM * 64 + mi * 16 + lhalf;      /* A 16x32 bf16: lanes<16 K0-7/16-23 */
        const bf16_t* ph = S + XH_OFF + row * LDSTR + lsel * 8;
        const bf16_t* pl = S + XL_OFF + row * LDSTR + lsel * 8;
        ah[mi] = mkfrag(ph, ph + 16);
        al[mi] = mkfrag(pl, pl + 16);
    }
#pragma unroll
    for (int ni = 0; ni < 4; ++ni) {
        int row = waveN * 64 + ni * 16 + lhalf;      /* B 32x16: lanes<16 K0-7/16-23, >=16 K8-15/24-31 */
        const bf16_t* ph = S + WH_OFF + row * LDSTR + lsel * 8;
        const bf16_t* pl = S + WL_OFF + row * LDSTR + lsel * 8;
        v16bf bh = mkfrag(ph, ph + 16);
        v16bf bl = mkfrag(pl, pl + 16);
#pragma unroll
        for (int mi = 0; mi < 4; ++mi) {
            c[mi][ni] = __builtin_amdgcn_wmma_f32_16x16x32_bf16(false, ah[mi], false, bh,
                                                                (short)0, c[mi][ni], false, false);
            c[mi][ni] = __builtin_amdgcn_wmma_f32_16x16x32_bf16(false, al[mi], false, bh,
                                                                (short)0, c[mi][ni], false, false);
            c[mi][ni] = __builtin_amdgcn_wmma_f32_16x16x32_bf16(false, ah[mi], false, bl,
                                                                (short)0, c[mi][ni], false, false);
        }
    }
}
#endif /* device */

__global__ void __launch_bounds__(256)
lora_gemm_kernel(const bf16_t* __restrict__ xh, const bf16_t* __restrict__ xl,
                 const bf16_t* __restrict__ wh, const bf16_t* __restrict__ wl,
                 const float* __restrict__ bias, float* __restrict__ out) {
#if defined(__HIP_DEVICE_COMPILE__)
    extern __shared__ bf16_t lds[];
    const size_t m0 = (size_t)blockIdx.y * 128;
    const size_t n0 = (size_t)blockIdx.x * 256;
    const int lane = threadIdx.x & 31;
    const int wave = threadIdx.x >> 5;
    const int waveM = wave >> 2, waveN = wave & 3;
    const int lhalf = lane & 15, lsel = lane >> 4;

    v8f c[4][4];
#pragma unroll
    for (int i = 0; i < 4; ++i)
#pragma unroll
        for (int j = 0; j < 4; ++j) c[i][j] = (v8f)0.0f;

    /* ---- prologue: stage slab 0 ---- */
#if USE_ASYNC_LDS
#pragma unroll
    for (int j = 0; j < 12; ++j) {
        int cc = threadIdx.x + j * 256;
        const bf16_t* g; int l;
        stage_addrs(cc, 0, m0, n0, xh, xl, wh, wl, &g, &l);
        __builtin_amdgcn_global_load_async_to_lds_b128((v4i*)g, (v4i*)(lds + l), 0, 0);
    }
    wait_async0();
#else
#pragma unroll
    for (int j = 0; j < 12; ++j) {
        int cc = threadIdx.x + j * 256;
        const bf16_t* g; int l;
        stage_addrs(cc, 0, m0, n0, xh, xl, wh, wl, &g, &l);
        *(v8bf*)(lds + l) = *(const v8bf*)g;
    }
#endif
    __syncthreads();

    /* ---- main K loop, double-buffered ---- */
    for (int kb = 0; kb < NKB; ++kb) {
        const bf16_t* cur = lds + (size_t)(kb & 1) * STAGE;
        bf16_t* nxt = lds + (size_t)((kb + 1) & 1) * STAGE;
#if USE_ASYNC_LDS
        if (kb + 1 < NKB) {
#pragma unroll
            for (int j = 0; j < 12; ++j) {
                int cc = threadIdx.x + j * 256;
                const bf16_t* g; int l;
                stage_addrs(cc, kb + 1, m0, n0, xh, xl, wh, wl, &g, &l);
                __builtin_amdgcn_global_load_async_to_lds_b128((v4i*)g, (v4i*)(nxt + l), 0, 0);
            }
        }
        compute_stage(cur, c, waveM, waveN, lhalf, lsel);
        if (kb + 1 < NKB) wait_async0();
        __syncthreads();
#else
        v8bf regs[12];
        if (kb + 1 < NKB) {
#pragma unroll
            for (int j = 0; j < 12; ++j) {
                int cc = threadIdx.x + j * 256;
                const bf16_t* g; int l;
                stage_addrs(cc, kb + 1, m0, n0, xh, xl, wh, wl, &g, &l);
                regs[j] = *(const v8bf*)g;
            }
        }
        compute_stage(cur, c, waveM, waveN, lhalf, lsel);
        if (kb + 1 < NKB) {
#pragma unroll
            for (int j = 0; j < 12; ++j) {
                int cc = threadIdx.x + j * 256;
                const bf16_t* g; int l;
                stage_addrs(cc, kb + 1, m0, n0, xh, xl, wh, wl, &g, &l);
                *(v8bf*)(nxt + l) = regs[j];
            }
        }
        __syncthreads();
#endif
    }

    /* ---- epilogue: bias + store (C layout: lane%16 = N col, vgpr j = row) ---- */
#pragma unroll
    for (int ni = 0; ni < 4; ++ni) {
        size_t col = n0 + (size_t)(waveN * 64 + ni * 16 + lhalf);
        float bv = bias[col];
#pragma unroll
        for (int mi = 0; mi < 4; ++mi) {
            size_t mbase = m0 + (size_t)(waveM * 64 + mi * 16 + lsel * 8);
            v8f cc = c[mi][ni];
#pragma unroll
            for (int j = 0; j < 8; ++j)
                out[(mbase + j) * (size_t)NDIM + col] = cc[j] + bv;
        }
    }
#endif
}

/* ------------------------------------------------------------------ */
extern "C" void kernel_launch(void* const* d_in, const int* in_sizes, int n_in,
                              void* d_out, int out_size, void* d_ws, size_t ws_size,
                              hipStream_t stream) {
    const float* x  = (const float*)d_in[0];
    const float* W  = (const float*)d_in[1];
    const float* bb = (const float*)d_in[2];
    const float* A  = (const float*)d_in[3];
    const float* Bm = (const float*)d_in[4];
    float* out = (float*)d_out;

    /* ws: xh | xl | wh | wl  (bf16)  = 192 MiB total */
    bf16_t* xh = (bf16_t*)d_ws;
    bf16_t* xl = xh + (size_t)MDIM * KDIM;
    bf16_t* wh = xl + (size_t)MDIM * KDIM;
    bf16_t* wl = wh + (size_t)NDIM * KDIM;

    int n4 = (MDIM * KDIM) / 4;
    splitx_kernel<<<n4 / 256, 256, 0, stream>>>(x, xh, xl, n4);
    weff_kernel<<<dim3(KDIM / 64, NDIM / 64), 256, 0, stream>>>(W, A, Bm, wh, wl);
    lora_gemm_kernel<<<dim3(NDIM / 256, MDIM / 128), 256, LDS_BYTES, stream>>>(
        xh, xl, wh, wl, bb, out);
}
